// Attention_44057774522397
// MI455X (gfx1250) — matrix operand order repaired and checked by reading, not compiled, not run on hardware
//
#include <hip/hip_runtime.h>
#include <math.h>

#define NUM_HEADS    32
#define NUM_KV_HEADS 8
#define GQA          4
#define HEAD_DIM     128
#define QK_SCALE     0.08838834764831845f  // 1/sqrt(128)

typedef __attribute__((ext_vector_type(16))) _Float16 v16h;
typedef __attribute__((ext_vector_type(8)))  _Float16 v8h;
typedef __attribute__((ext_vector_type(4)))  _Float16 v4h;
typedef __attribute__((ext_vector_type(8)))  float    v8f;

// LDS row strides (f16 elements); chosen so every v8h (16B) access is 16B
// aligned: 136*2=272B and 40*2=80B are multiples of 16.
#define QS 136
#define KS 136
#define VS 40
#define PS 40

__device__ __forceinline__ v16h cat8(v8h lo, v8h hi) {
  return __builtin_shufflevector(lo, hi, 0,1,2,3,4,5,6,7,8,9,10,11,12,13,14,15);
}
__device__ __forceinline__ v16h ld_b16(const _Float16* p) {
  v8h lo = *(const v8h*)(p);
  v8h hi = *(const v8h*)(p + 8);
  return cat8(lo, hi);
}

// ---------------------------------------------------------------------------
// Flash-attention prefill: one block = (q-tile of 16 rows, one KV head).
// 4 waves/block; wave w computes Q-head kvh*4+w. K/V blocks double-buffered
// in LDS (f16); next block's global loads overlap current block's WMMAs.
// ---------------------------------------------------------------------------
__global__ __launch_bounds__(128)
void fa_prefill(const float* __restrict__ q,
                const float* __restrict__ k,
                const float* __restrict__ v,
                const int*   __restrict__ cu,
                float*       __restrict__ out,
                int nB)
{
  __shared__ alignas(16) _Float16 sQ[GQA][16][QS];        // f16 Q (pre-scaled)
  __shared__ alignas(16) _Float16 sK[2][32][KS];          // K block, row-major
  __shared__ alignas(16) _Float16 sVt[2][HEAD_DIM][VS];   // V block, transposed
  __shared__ alignas(16) _Float16 sP[GQA][16][PS];        // per-wave P scratch

  const int kvh = blockIdx.y;

  // decode (batch, tile) from flat tile index via cu_seqlens scan
  int rem = blockIdx.x;
  int b = -1, tile = 0, q0 = 0, sl = 0;
  for (int i = 0; i < nB; ++i) {
    int s = cu[i], e = cu[i + 1];
    int len = e - s;
    int nt = (len + 15) >> 4;
    if (rem < nt) { b = i; tile = rem; q0 = s; sl = len; break; }
    rem -= nt;
  }
  if (b < 0) return;

  const int tid  = threadIdx.x;
  const int wid  = tid >> 5;          // which Q head of the GQA group
  const int lane = tid & 31;
  const int nidx = lane & 15;         // N index in B/C/D fragments
  const int half = lane >> 4;         // lane half (K / M split)
  const int head = kvh * GQA + wid;
  const int qtok0 = q0 + tile * 16;
  const int qend  = q0 + sl;

  // ---- stage Q for all 4 heads of this group (scale folded in) ----
  for (int idx = tid; idx < GQA * 16 * HEAD_DIM; idx += 128) {
    int g = idx >> 11;
    int r = idx & 2047;
    int m = r >> 7;
    int d = r & 127;
    int tok = qtok0 + m;
    float val = (tok < qend)
        ? q[((size_t)tok * NUM_HEADS + (kvh * GQA + g)) * HEAD_DIM + d] * QK_SCALE
        : 0.f;
    sQ[g][m][d] = (_Float16)val;
  }

  // ---- per-thread K/V staging registers: 8 float4 -> 8 v4h each (32 VGPRs) ----
  v4h stK[8], stV[8];
  const int kend = min(sl, tile * 16 + 16);  // causal: keys needed by last row

  auto load_stage = [&](int kb) {
#pragma unroll
    for (int j = 0; j < 8; ++j) {
      int f4   = tid + j * 128;      // float4 index in the 32x128 block
      int base = f4 << 2;
      int key  = base >> 7;
      int d    = base & 127;
      int kt   = kb + key;
      float4 kf = make_float4(0.f, 0.f, 0.f, 0.f);
      float4 vf = make_float4(0.f, 0.f, 0.f, 0.f);
      if (kt < sl) {
        size_t g = ((size_t)(q0 + kt) * NUM_KV_HEADS + kvh) * HEAD_DIM + d;
        kf = *(const float4*)(k + g);
        vf = *(const float4*)(v + g);
      }
      v4h kh, vh;
      kh[0] = (_Float16)kf.x; kh[1] = (_Float16)kf.y;
      kh[2] = (_Float16)kf.z; kh[3] = (_Float16)kf.w;
      vh[0] = (_Float16)vf.x; vh[1] = (_Float16)vf.y;
      vh[2] = (_Float16)vf.z; vh[3] = (_Float16)vf.w;
      stK[j] = kh; stV[j] = vh;
    }
  };
  auto store_stage = [&](int buf) {
#pragma unroll
    for (int j = 0; j < 8; ++j) {
      int f4   = tid + j * 128;
      int base = f4 << 2;
      int key  = base >> 7;
      int d    = base & 127;
      *(v4h*)&sK[buf][key][d] = stK[j];            // one b64 store
      sVt[buf][d + 0][key] = stV[j][0];            // transposed scatter
      sVt[buf][d + 1][key] = stV[j][1];
      sVt[buf][d + 2][key] = stV[j][2];
      sVt[buf][d + 3][key] = stV[j][3];
    }
  };

  load_stage(0);           // prologue: fetch first K/V block (overlaps Q cvt)
  __syncthreads();         // Q staged

  // ---- Q A-fragments, resident in VGPRs for the whole K loop ----
  // A layout (16x32 f16): half0 lanes hold K 0..15, half1 K 16..31 (matches B)
  v16h qa[4];
  {
    const _Float16* row = &sQ[wid][nidx][0];
    const int koff = half * 16;
#pragma unroll
    for (int c = 0; c < 4; ++c) {
      int k0 = c * 32;
      v8h lo = *(const v8h*)(row + k0 + koff);
      v8h hi = *(const v8h*)(row + k0 + 8 + koff);
      qa[c] = cat8(lo, hi);
    }
  }

  v8f zero = {};
  v8f acc[8];
#pragma unroll
  for (int i = 0; i < 8; ++i) acc[i] = zero;
  float mi[8], li[8];
#pragma unroll
  for (int r = 0; r < 8; ++r) { mi[r] = -1.0e30f; li[r] = 0.f; }

  int cur = 0;
  for (int kb = 0; kb < kend; kb += 32) {
    store_stage(cur);      // regs -> LDS buf[cur]
    __syncthreads();       // one barrier per iteration (double buffered)
    if (kb + 32 < kend) load_stage(kb + 32);   // prefetch next block early

    // ---- S = Q @ K^T : batch all 8 B-fragments, then 8 dense WMMAs ----
    v16h bf[8];
#pragma unroll
    for (int c = 0; c < 4; ++c) {
      int k0 = c * 32 + half * 16;   // B layout: lanes 0-15 K 0..15, 16-31 K 16..31
      bf[2 * c]     = ld_b16(&sK[cur][nidx][k0]);
      bf[2 * c + 1] = ld_b16(&sK[cur][16 + nidx][k0]);
    }
    v8f s0 = zero, s1 = zero;
#pragma unroll
    for (int c = 0; c < 4; ++c) {
      s0 = __builtin_amdgcn_wmma_f32_16x16x32_f16(false, qa[c], false, bf[2 * c],
                                                  (short)0, s0, false, false);
      s1 = __builtin_amdgcn_wmma_f32_16x16x32_f16(false, qa[c], false, bf[2 * c + 1],
                                                  (short)0, s1, false, false);
    }

    // ---- issue P@V B-fragment loads now; they overlap the softmax VALU ----
    v16h bv[8];
#pragma unroll
    for (int nt = 0; nt < 8; ++nt)
      bv[nt] = ld_b16(&sVt[cur][nt * 16 + nidx][half * 16]);

    // ---- online softmax (row stats per D-layout row: M = r + 8*half) ----
#pragma unroll
    for (int r = 0; r < 8; ++r) {
      int qm  = tile * 16 + r + 8 * half;  // query pos within sequence
      int kp0 = kb + nidx;
      int kp1 = kb + 16 + nidx;
      float e0 = (kp0 <= qm) ? s0[r] : -1.0e30f;
      float e1 = (kp1 <= qm) ? s1[r] : -1.0e30f;
      float mx = fmaxf(e0, e1);
#pragma unroll
      for (int msk = 1; msk < 16; msk <<= 1)       // 16-lane row reduce
        mx = fmaxf(mx, __shfl_xor(mx, msk, 32));
      float mnew = fmaxf(mi[r], mx);
      float p0 = __expf(e0 - mnew);
      float p1 = __expf(e1 - mnew);
      float rs = p0 + p1;
#pragma unroll
      for (int msk = 1; msk < 16; msk <<= 1)
        rs += __shfl_xor(rs, msk, 32);
      float alpha = __expf(mi[r] - mnew);
      li[r] = li[r] * alpha + rs;
      mi[r] = mnew;
#pragma unroll
      for (int nt = 0; nt < 8; ++nt) acc[nt][r] *= alpha;
      int M = r + 8 * half;
      sP[wid][M][nidx]      = (_Float16)p0;   // D-layout -> LDS (row-major)
      sP[wid][M][16 + nidx] = (_Float16)p1;
    }

    // ---- O += P @ V (P re-read in A layout; per-wave DS is in-order) ----
    {
      const _Float16* prow = &sP[wid][nidx][0];
      const int koff = half * 16;
      v8h lo = *(const v8h*)(prow + koff);
      v8h hi = *(const v8h*)(prow + 8 + koff);
      v16h pa = cat8(lo, hi);
#pragma unroll
      for (int nt = 0; nt < 8; ++nt)
        acc[nt] = __builtin_amdgcn_wmma_f32_16x16x32_f16(false, pa, false, bv[nt],
                                                         (short)0, acc[nt], false, false);
    }
    cur ^= 1;
  }

  // ---- epilogue: O /= l, store valid rows (lanes 0-15 share a row => 64B runs) ----
#pragma unroll
  for (int r = 0; r < 8; ++r) {
    int tok = qtok0 + r + 8 * half;
    if (tok < qend) {
      float inv = 1.0f / li[r];
      size_t base = ((size_t)tok * NUM_HEADS + head) * HEAD_DIM;
#pragma unroll
      for (int nt = 0; nt < 8; ++nt)
        out[base + nt * 16 + nidx] = acc[nt][r] * inv;
    }
  }
}

// ---------------------------------------------------------------------------
// Cache handling: outputs are full caches, inputs must not be mutated, so
// copy inputs -> output region, then scatter new K/V by slot_mapping.
// ---------------------------------------------------------------------------
__global__ void cache_copy(const float* __restrict__ kc, const float* __restrict__ vc,
                           float* __restrict__ okc, float* __restrict__ ovc, size_t n)
{
  size_t i = (size_t)blockIdx.x * blockDim.x + threadIdx.x;
  size_t stride = (size_t)gridDim.x * blockDim.x;
  for (; i < n; i += stride) { okc[i] = kc[i]; ovc[i] = vc[i]; }
}

__global__ void kv_scatter(const float* __restrict__ k, const float* __restrict__ v,
                           const int* __restrict__ slot,
                           float* __restrict__ okc, float* __restrict__ ovc, int T)
{
  int tok = blockIdx.x;
  if (tok >= T) return;
  int s = slot[tok];
  if (s < 0) return;
  const int row = NUM_KV_HEADS * HEAD_DIM;  // 1024 floats per token
  size_t src = (size_t)tok * row;
  size_t dst = (size_t)s * row;
  for (int i = threadIdx.x; i < row; i += blockDim.x) {
    okc[dst + i] = k[src + i];
    ovc[dst + i] = v[src + i];
  }
}

extern "C" void kernel_launch(void* const* d_in, const int* in_sizes, int n_in,
                              void* d_out, int out_size, void* d_ws, size_t ws_size,
                              hipStream_t stream)
{
  const float* q    = (const float*)d_in[0];
  const float* k    = (const float*)d_in[1];
  const float* v    = (const float*)d_in[2];
  const float* kc   = (const float*)d_in[3];
  const float* vc   = (const float*)d_in[4];
  const int*   cu   = (const int*)d_in[5];
  const int*   slot = (const int*)d_in[6];

  const int T  = in_sizes[0] / (NUM_HEADS * HEAD_DIM);
  const int nB = in_sizes[5] - 1;
  const size_t cacheN = (size_t)in_sizes[3];

  float* out = (float*)d_out;
  float* okc = out + (size_t)T * NUM_HEADS * HEAD_DIM;
  float* ovc = okc + cacheN;

  // caches: copy then scatter (stream-ordered)
  {
    int threads = 256;
    int blocks = (int)((cacheN + threads - 1) / threads);
    if (blocks > 65535) blocks = 65535;  // grid-stride loop covers the rest
    cache_copy<<<blocks, threads, 0, stream>>>(kc, vc, okc, ovc, cacheN);
    kv_scatter<<<T, 256, 0, stream>>>(k, v, slot, okc, ovc, T);
  }

  // attention: upper bound on total q-tiles = T/16 + nB (per-seq rounding)
  {
    int maxTiles = T / 16 + nB;
    dim3 grid(maxTiles, NUM_KV_HEADS);
    fa_prefill<<<grid, 128, 0, stream>>>(q, k, v, cu, out, nB);
  }
}
